// SpatialMambaBlock_22076131902166
// MI455X (gfx1250) — matrix operand order repaired and checked by reading, compiled, not run on hardware
//
#include <hip/hip_runtime.h>
#include <hip/hip_bf16.h>

// ---------------- problem constants ----------------
#define BT      48          // B*T
#define NSEQ    1024        // nodes per sequence
#define DMODEL  128
#define DINNER  256
#define DSTATE  16
#define DTRANK  8
#define ROWS    (BT * NSEQ) // 49152
#define XDBL_LD 48          // dt_rank + 2*d_state = 40, padded to 48

typedef __attribute__((ext_vector_type(16))) __bf16 v16bf;
typedef __attribute__((ext_vector_type(8)))  __bf16 v8bf;
typedef __attribute__((ext_vector_type(8)))  float  v8f;

__device__ __forceinline__ unsigned short f2bf(float f) {
    union { float f; unsigned u; } v; v.f = f;
    unsigned r = v.u + 0x7FFFu + ((v.u >> 16) & 1u);
    return (unsigned short)(r >> 16);
}
__device__ __forceinline__ float bf2f(unsigned short h) {
    union { unsigned u; float f; } v; v.u = ((unsigned)h) << 16;
    return v.f;
}

// ---------------- weight fp32 -> bf16 conversion (+pad x_proj to 48 rows) ----
__global__ void k_convert_weights(const float* __restrict__ w1,   // 512x128
                                  const float* __restrict__ w2,   // 40x256
                                  const float* __restrict__ w3,   // 128x256
                                  unsigned short* __restrict__ o1,
                                  unsigned short* __restrict__ o2, // 48x256
                                  unsigned short* __restrict__ o3) {
    int idx = blockIdx.x * 256 + threadIdx.x;
    if (idx < 512 * 128) { o1[idx] = f2bf(w1[idx]); return; }
    idx -= 512 * 128;
    if (idx < 48 * 256) {
        int r = idx >> 8, k = idx & 255;
        o2[idx] = (r < 40) ? f2bf(w2[r * 256 + k]) : (unsigned short)0;
        return;
    }
    idx -= 48 * 256;
    if (idx < 128 * 256) o3[idx] = f2bf(w3[idx]);
}

// ---------------- LayerNorm -> bf16 (one wave32 per row of 128) -------------
__global__ void k_layernorm(const float* __restrict__ x,
                            const float* __restrict__ lnw,
                            const float* __restrict__ lnb,
                            unsigned short* __restrict__ xn) {
    int wave = threadIdx.x >> 5, lane = threadIdx.x & 31;
    int row  = blockIdx.x * 8 + wave;
    const float4 v = *(const float4*)(x + (size_t)row * DMODEL + lane * 4);
    float s = v.x + v.y + v.z + v.w;
    float q = v.x * v.x + v.y * v.y + v.z * v.z + v.w * v.w;
    #pragma unroll
    for (int off = 16; off >= 1; off >>= 1) {
        s += __shfl_xor(s, off);
        q += __shfl_xor(q, off);
    }
    float mean = s * (1.0f / DMODEL);
    float var  = q * (1.0f / DMODEL) - mean * mean;
    float rstd = rsqrtf(var + 1e-5f);
    const float4 gw = *(const float4*)(lnw + lane * 4);
    const float4 gb = *(const float4*)(lnb + lane * 4);
    ushort4 o;
    o.x = f2bf((v.x - mean) * rstd * gw.x + gb.x);
    o.y = f2bf((v.y - mean) * rstd * gw.y + gb.y);
    o.z = f2bf((v.z - mean) * rstd * gw.z + gb.z);
    o.w = f2bf((v.w - mean) * rstd * gw.w + gb.w);
    *(ushort4*)(xn + (size_t)row * DMODEL + lane * 4) = o;
}

// ---------------- WMMA bf16 GEMM:  C[M,N] = A[M,K] * W[N,K]^T ---------------
// MODE 0: store bf16    MODE 1: store fp32    MODE 2: store fp32 + residual
template <int MODE>
__global__ void k_gemm_wmma(const __bf16* __restrict__ A, int lda,
                            const __bf16* __restrict__ W, int ldw,
                            void* __restrict__ Cout, int ldc,
                            const float* __restrict__ resid,
                            int Mtiles, int Ntiles, int K) {
    int wave = threadIdx.x >> 5;
    int lane = threadIdx.x & 31;
    int tile = blockIdx.x * 8 + wave;
    if (tile >= Mtiles * Ntiles) return;          // wave-uniform exit: EXEC stays full
    int mt = tile / Ntiles, nt = tile % Ntiles;
    int half = lane >> 4, l = lane & 15;

    const __bf16* arow = A + (size_t)(mt * 16 + l) * lda;
    const __bf16* wrow = W + (size_t)(nt * 16 + l) * ldw;

    v8f c = {};
    for (int kc = 0; kc < K; kc += 32) {
        // A 16x32 bf16: lane<16 -> K {kc..kc+7, kc+16..kc+23}; lane>=16 -> +8
        v8bf a0 = *(const v8bf*)(arow + kc + half * 8);
        v8bf a1 = *(const v8bf*)(arow + kc + 16 + half * 8);
        // B 32x16 bf16 (= W^T): lane<16 -> K {kc..kc+7, kc+16..kc+23}; lane>=16 -> +8
        v8bf b0 = *(const v8bf*)(wrow + kc + half * 8);
        v8bf b1 = *(const v8bf*)(wrow + kc + 16 + half * 8);
        v16bf av, bv;
        #pragma unroll
        for (int i = 0; i < 8; ++i) {
            av[i] = a0[i]; av[i + 8] = a1[i];
            bv[i] = b0[i]; bv[i + 8] = b1[i];
        }
        c = __builtin_amdgcn_wmma_f32_16x16x32_bf16(false, av, false, bv,
                                                    (short)0, c, false, false);
    }

    int ocol = nt * 16 + l;
    #pragma unroll
    for (int r = 0; r < 8; ++r) {
        int orow = mt * 16 + r + half * 8;        // C layout: VGPR r -> M=r (+8 upper half)
        float v = c[r];
        if (MODE == 0) {
            ((unsigned short*)Cout)[(size_t)orow * ldc + ocol] = f2bf(v);
        } else if (MODE == 1) {
            ((float*)Cout)[(size_t)orow * ldc + ocol] = v;
        } else {
            ((float*)Cout)[(size_t)orow * ldc + ocol] =
                v + resid[(size_t)orow * ldc + ocol];
        }
    }
}

// ---------------- depthwise causal conv (k=4) + bias + SiLU -----------------
__global__ void k_conv_silu(const unsigned short* __restrict__ xz,  // ROWS x 512 bf16
                            const float* __restrict__ cw,           // 256x4
                            const float* __restrict__ cb,
                            float* __restrict__ xcf,                // ROWS x 256 f32
                            unsigned short* __restrict__ xcb) {     // ROWS x 256 bf16
    int idx = blockIdx.x * 256 + threadIdx.x;
    int c = idx & 255, row = idx >> 8, n = row & (NSEQ - 1);
    float acc = cb[c];
    #pragma unroll
    for (int j = 0; j < 4; ++j) {
        int m = n - 3 + j;
        if (m >= 0)
            acc += cw[c * 4 + j] * bf2f(xz[(size_t)(row - 3 + j) * 512 + c]);
    }
    float sv = acc / (1.0f + __expf(-acc));       // SiLU
    xcf[(size_t)row * DINNER + c] = sv;
    xcb[(size_t)row * DINNER + c] = f2bf(sv);
}

// ---------------- dt = softplus(dt_in @ dt_proj_w^T + b) --------------------
__global__ void k_dt(const float* __restrict__ xdbl,     // ROWS x 48
                     const float* __restrict__ dpw,      // 256x8
                     const float* __restrict__ dpb,
                     float* __restrict__ dt) {
    int idx = blockIdx.x * 256 + threadIdx.x;
    int d = idx & 255, row = idx >> 8;
    float acc = dpb[d];
    #pragma unroll
    for (int r = 0; r < DTRANK; ++r)
        acc += xdbl[(size_t)row * XDBL_LD + r] * dpw[d * DTRANK + r];
    float sp = (acc > 20.0f) ? acc : log1pf(__expf(acc));
    dt[(size_t)row * DINNER + d] = sp;
}

// ---------------- selective scan (16 lanes per channel, scalar h) -----------
// fuses: y = (scan + D*u) * silu(z)  -> bf16
__global__ void k_scan(const float* __restrict__ dt,
                       const float* __restrict__ xcf,
                       const float* __restrict__ xdbl,
                       const unsigned short* __restrict__ xz,   // z at cols 256..511
                       const float* __restrict__ A_log,
                       const float* __restrict__ Dskip,
                       unsigned short* __restrict__ ybf) {
    int b = blockIdx.x >> 4;                              // 48 batches x 16 blocks
    int d = ((blockIdx.x & 15) << 4) + (threadIdx.x >> 4);
    int s = threadIdx.x & 15;
    float Aneg = -__expf(A_log[d * DSTATE + s]);
    float Dsk  = Dskip[d];
    float h = 0.0f;
    size_t rowBase = (size_t)b * NSEQ;
    for (int n = 0; n < NSEQ; ++n) {
        size_t row = rowBase + n;
        float dti = dt[row * DINNER + d];                 // broadcast in 16-lane group
        float u   = xcf[row * DINNER + d];
        float Bv  = xdbl[row * XDBL_LD + DTRANK + s];
        float Cv  = xdbl[row * XDBL_LD + DTRANK + DSTATE + s];
        float dA  = __expf(dti * Aneg);                   // 1 transcendental / lane / step
        h = dA * h + (dti * u) * Bv;
        float p = h * Cv;
        p += __shfl_xor(p, 1, 16);
        p += __shfl_xor(p, 2, 16);
        p += __shfl_xor(p, 4, 16);
        p += __shfl_xor(p, 8, 16);
        if (s == 0) {
            float zv = bf2f(xz[row * 512 + 256 + d]);
            float g  = zv / (1.0f + __expf(-zv));
            ybf[row * DINNER + d] = f2bf((p + Dsk * u) * g);
        }
    }
}

// ---------------- launcher ----------------
extern "C" void kernel_launch(void* const* d_in, const int* in_sizes, int n_in,
                              void* d_out, int out_size, void* d_ws, size_t ws_size,
                              hipStream_t stream) {
    const float* x      = (const float*)d_in[0];
    const float* ln_w   = (const float*)d_in[1];
    const float* ln_b   = (const float*)d_in[2];
    const float* w_in   = (const float*)d_in[3];   // 512x128
    const float* conv_w = (const float*)d_in[4];   // 256x1x4
    const float* conv_b = (const float*)d_in[5];
    const float* w_xp   = (const float*)d_in[6];   // 40x256
    const float* w_dt   = (const float*)d_in[7];   // 256x8
    const float* b_dt   = (const float*)d_in[8];
    const float* A_log  = (const float*)d_in[9];   // 256x16
    const float* D_skip = (const float*)d_in[10];
    const float* w_out  = (const float*)d_in[11];  // 128x256

    char* ws = (char*)d_ws;
    size_t off = 0;
    unsigned short* xn   = (unsigned short*)(ws + off); off += (size_t)ROWS * DMODEL * 2;   // 12.6MB
    unsigned short* w1b  = (unsigned short*)(ws + off); off += 512 * 128 * 2;
    unsigned short* w2b  = (unsigned short*)(ws + off); off += 48 * 256 * 2;
    unsigned short* w3b  = (unsigned short*)(ws + off); off += 128 * 256 * 2;
    unsigned short* xz   = (unsigned short*)(ws + off); off += (size_t)ROWS * 512 * 2;      // 50.3MB
    float*          xcf  = (float*)        (ws + off); off += (size_t)ROWS * DINNER * 4;    // 50.3MB
    unsigned short* xcb  = (unsigned short*)(ws + off); off += (size_t)ROWS * DINNER * 2;   // 25.2MB
    float*          xdbl = (float*)        (ws + off); off += (size_t)ROWS * XDBL_LD * 4;   // 9.4MB
    float*          dt   = (float*)        (ws + off); off += (size_t)ROWS * DINNER * 4;    // 50.3MB
    unsigned short* ybf  = (unsigned short*)(ws + off); off += (size_t)ROWS * DINNER * 2;   // 25.2MB

    // 1. weights -> bf16 (x_proj padded 40->48 rows)
    k_convert_weights<<<(512 * 128 + 48 * 256 + 128 * 256 + 255) / 256, 256, 0, stream>>>(
        w_in, w_xp, w_out, w1b, w2b, w3b);

    // 2. LayerNorm -> bf16
    k_layernorm<<<ROWS / 8, 256, 0, stream>>>(x, ln_w, ln_b, xn);

    // 3. in_proj GEMM: (49152x128)x(128x512) -> xz bf16
    k_gemm_wmma<0><<<(ROWS / 16) * (512 / 16) / 8, 256, 0, stream>>>(
        (const __bf16*)xn, DMODEL, (const __bf16*)w1b, DMODEL,
        (void*)xz, 512, nullptr, ROWS / 16, 512 / 16, DMODEL);

    // 4. depthwise conv + SiLU
    k_conv_silu<<<ROWS * DINNER / 256, 256, 0, stream>>>(xz, conv_w, conv_b, xcf, xcb);

    // 5. x_proj GEMM: (49152x256)x(256x48) -> x_dbl f32
    k_gemm_wmma<1><<<(ROWS / 16) * (XDBL_LD / 16) / 8, 256, 0, stream>>>(
        (const __bf16*)xcb, DINNER, (const __bf16*)w2b, DINNER,
        (void*)xdbl, XDBL_LD, nullptr, ROWS / 16, XDBL_LD / 16, DINNER);

    // 6. dt projection + softplus
    k_dt<<<ROWS * DINNER / 256, 256, 0, stream>>>(xdbl, w_dt, b_dt, dt);

    // 7. selective scan + D-skip + SiLU(z) gate -> bf16
    k_scan<<<BT * (DINNER / 16), 256, 0, stream>>>(dt, xcf, xdbl, xz, A_log, D_skip, ybf);

    // 8. out_proj GEMM + residual: (49152x256)x(256x128) + x -> d_out f32
    k_gemm_wmma<2><<<(ROWS / 16) * (DMODEL / 16) / 8, 256, 0, stream>>>(
        (const __bf16*)ybf, DINNER, (const __bf16*)w3b, DINNER,
        d_out, DMODEL, x, ROWS / 16, DMODEL / 16, DINNER);

    (void)in_sizes; (void)n_in; (void)out_size; (void)ws_size;
}
